// TumorTemporalGNN_34557306864290
// MI455X (gfx1250) — matrix operand order repaired and checked by reading, not compiled, not run on hardware
//
#include <hip/hip_runtime.h>
#include <hip/hip_fp16.h>
#include <math.h>

// ---------------- problem constants (match reference) ----------------
#define GCNT   128     // B*T graphs
#define NNODE  512
#define NEDGE  4096
#define DMODEL 128     // H*CH
#define NHEAD  4
#define CHEAD  32
#define TSEQ   8
#define BATCH  16
#define NCLS   3

typedef __attribute__((ext_vector_type(16))) _Float16 v16h;
typedef __attribute__((ext_vector_type(8)))  _Float16 v8h;
typedef __attribute__((ext_vector_type(8)))  float    v8f;

// ---------------------------------------------------------------------
// f32 -> f16 streaming convert (grid-stride)
// ---------------------------------------------------------------------
__global__ __launch_bounds__(256) void f32_to_f16(
    const float* __restrict__ in, _Float16* __restrict__ out, int n)
{
    for (int i = blockIdx.x * blockDim.x + threadIdx.x; i < n;
         i += gridDim.x * blockDim.x)
        out[i] = (_Float16)in[i];
}

// ---------------------------------------------------------------------
// weight transpose + convert: WT[n][k] = (f16) W[k][n], W is [K,128]
// grid = K blocks, block = 128 threads
// ---------------------------------------------------------------------
__global__ __launch_bounds__(128) void wtrans(
    const float* __restrict__ W, _Float16* __restrict__ WT, int K)
{
    const int k = blockIdx.x, n = threadIdx.x;
    WT[(size_t)n * K + k] = (_Float16)W[(size_t)k * DMODEL + n];
}

// ---------------------------------------------------------------------
// WMMA GEMM: OUT[g, 0:nrows, 0:128] = XH[g, 0:nrows, 0:K] @ W (f16),
// with the weight stored transposed: WT[n][k], n = 0..127, k = 0..K-1.
// One wave per 16-row M-tile, 8 N-tile accumulators (full 128-wide strip).
// All fragment loads are 16-byte vector loads (global_load_b128).
// block = 128 threads (4 waves), grid = (G, nrows/64). K % 32 == 0.
// ---------------------------------------------------------------------
__global__ __launch_bounds__(128) void wmma_gemm128(
    const _Float16* __restrict__ XH, const _Float16* __restrict__ WT,
    float* __restrict__ OUT, int nrows, int K)
{
    const int g    = blockIdx.x;
    const int wave = threadIdx.x >> 5;
    const int lane = threadIdx.x & 31;
    const int mtile = blockIdx.y * 4 + wave;
    if (mtile * 16 >= nrows) return;          // uniform per-wave

    const _Float16* Xg = XH + (size_t)g * nrows * K;
    float*          Og = OUT + (size_t)g * nrows * DMODEL;

    const int lo = lane & 15;
    const int hi = lane >> 4;
    const int mrow = mtile * 16 + lo;

    v8f acc[8] = {};

    __builtin_prefetch(WT, 0, 3);             // global_prefetch of weights

    for (int kk = 0; kk < K; kk += 32) {
        // A fragment (16x32 f16): elems 0-7 = K kb..kb+7, 8-15 = kb+16..kb+23
        const _Float16* xr = Xg + (size_t)mrow * K + kk + hi * 8;
        v8h a0 = *reinterpret_cast<const v8h*>(xr);
        v8h a1 = *reinterpret_cast<const v8h*>(xr + 16);
        v16h a = __builtin_shufflevector(a0, a1,
                     0,1,2,3,4,5,6,7,8,9,10,11,12,13,14,15);
        #pragma unroll
        for (int nt = 0; nt < 8; ++nt) {
            // B fragment (32x16 f16): col = lo, elems 0-7 = K kb..kb+7, 8-15 = kb+16..kb+23
            const _Float16* wr = WT + (size_t)(nt * 16 + lo) * K + kk + hi * 8;
            v8h b0 = *reinterpret_cast<const v8h*>(wr);
            v8h b1 = *reinterpret_cast<const v8h*>(wr + 16);
            v16h bf = __builtin_shufflevector(b0, b1,
                          0,1,2,3,4,5,6,7,8,9,10,11,12,13,14,15);
            acc[nt] = __builtin_amdgcn_wmma_f32_16x16x32_f16(
                false, a, false, bf, (short)0, acc[nt], false, false);
        }
    }

    #pragma unroll
    for (int nt = 0; nt < 8; ++nt) {
        #pragma unroll
        for (int r = 0; r < 8; ++r) {
            const int row = mtile * 16 + hi * 8 + r;
            const int col = nt * 16 + lo;
            Og[(size_t)row * DMODEL + col] = acc[nt][r];
        }
    }
}

// ---------------------------------------------------------------------
// GAT attention + aggregation, one 512-thread block per graph.
// Builds a CSR of incoming edges in LDS (count -> scan -> fill),
// then each thread GATHERS its node's edges per head: max, sum, weighted
// accumulate in registers. No global atomics at all.
// Writes f32 output and (optionally) an f16 copy for the next WMMA GEMM.
// ---------------------------------------------------------------------
__global__ __launch_bounds__(512) void gat_attn(
    const float* __restrict__ HP, const float* __restrict__ EA,
    const int* __restrict__ EIDX, const float* __restrict__ We,
    const float* __restrict__ a_s, const float* __restrict__ a_d,
    const float* __restrict__ a_e, const float* __restrict__ bias,
    float* __restrict__ OUTF, _Float16* __restrict__ OUTH)
{
    __shared__ float ssrc[NNODE][NHEAD];     // 8 KB
    __shared__ float sdst[NNODE][NHEAD];     // 8 KB
    __shared__ int   cnt [NNODE];            // 2 KB (count, then cursor)
    __shared__ int   off [NNODE];            // 2 KB (scan / exclusive offsets)
    __shared__ int   elist[NEDGE];           // 16 KB (edge ids bucketed by dst)
    __shared__ float wae [2][NHEAD];

    const int g   = blockIdx.x;
    const int tid = threadIdx.x;             // 512 threads == NNODE
    const float* HPg = HP + (size_t)g * NNODE * DMODEL;
    const float* EAg = EA + (size_t)g * NEDGE * 2;
    float*       Og  = OUTF + (size_t)g * NNODE * DMODEL;

    if (tid < 8) {                            // fold a_e into We
        const int k = tid >> 2, h = tid & 3;
        float s = 0.f;
        for (int c = 0; c < CHEAD; ++c)
            s += We[k * DMODEL + h * CHEAD + c] * a_e[h * CHEAD + c];
        wae[k][h] = s;
    }

    {   // per-node attention scores
        const float* hr = HPg + (size_t)tid * DMODEL;
        #pragma unroll
        for (int h = 0; h < NHEAD; ++h) {
            float s1 = 0.f, s2 = 0.f;
            #pragma unroll
            for (int c = 0; c < CHEAD; ++c) {
                const float v = hr[h * CHEAD + c];
                s1 += v * a_s[h * CHEAD + c];
                s2 += v * a_d[h * CHEAD + c];
            }
            ssrc[tid][h] = s1; sdst[tid][h] = s2;
        }
    }

    // ---- CSR build: in-degree count ----
    cnt[tid] = 0;
    __syncthreads();
    for (int e = tid; e < NEDGE; e += NNODE)
        atomicAdd(&cnt[EIDX[NEDGE + e]], 1);
    __syncthreads();

    // ---- Hillis-Steele inclusive scan over 512 counts -> exclusive off ----
    const int myc = cnt[tid];
    off[tid] = myc;
    __syncthreads();
    for (int st = 1; st < NNODE; st <<= 1) {
        int v = off[tid];
        if (tid >= st) v += off[tid - st];
        __syncthreads();
        off[tid] = v;
        __syncthreads();
    }
    const int myoff = off[tid] - myc;        // exclusive prefix
    __syncthreads();
    off[tid] = myoff;
    cnt[tid] = 0;                            // reuse as fill cursor
    __syncthreads();

    // ---- CSR fill ----
    for (int e = tid; e < NEDGE; e += NNODE) {
        const int d = EIDX[NEDGE + e];
        const int p = off[d] + atomicAdd(&cnt[d], 1);
        elist[p] = e;
    }
    __syncthreads();

    // ---- per-node gather: softmax over incoming edges, weighted sum ----
    const int n    = tid;
    const int deg  = cnt[n];
    const int base = off[n];

    for (int h = 0; h < NHEAD; ++h) {
        const float w0 = wae[0][h], w1 = wae[1][h];
        const float sd = sdst[n][h];

        float m = -1e30f;
        for (int i = 0; i < deg; ++i) {
            const int e = elist[base + i];
            float l = ssrc[EIDX[e]][h] + sd
                    + EAg[e * 2] * w0 + EAg[e * 2 + 1] * w1;
            l = l > 0.f ? l : 0.2f * l;
            m = fmaxf(m, l);
        }
        float den = 0.f;
        for (int i = 0; i < deg; ++i) {
            const int e = elist[base + i];
            float l = ssrc[EIDX[e]][h] + sd
                    + EAg[e * 2] * w0 + EAg[e * 2 + 1] * w1;
            l = l > 0.f ? l : 0.2f * l;
            den += __expf(l - m);
        }
        const float dinv = 1.f / (den + 1e-16f);

        float accv[CHEAD];
        #pragma unroll
        for (int c = 0; c < CHEAD; ++c) accv[c] = 0.f;
        for (int i = 0; i < deg; ++i) {
            const int e = elist[base + i];
            const int s = EIDX[e];
            float l = ssrc[s][h] + sd
                    + EAg[e * 2] * w0 + EAg[e * 2 + 1] * w1;
            l = l > 0.f ? l : 0.2f * l;
            const float alpha = __expf(l - m) * dinv;
            const float* hs = HPg + (size_t)s * DMODEL + h * CHEAD;
            #pragma unroll
            for (int c = 0; c < CHEAD; ++c) accv[c] += alpha * hs[c];
        }
        #pragma unroll
        for (int c = 0; c < CHEAD; ++c) {
            float v = accv[c] + bias[h * CHEAD + c];
            v = v > 0.f ? v : (__expf(v) - 1.f);           // ELU
            const size_t oidx = (size_t)n * DMODEL + h * CHEAD + c;
            Og[oidx] = v;
            if (OUTH) OUTH[(size_t)g * NNODE * DMODEL + oidx] = (_Float16)v;
        }
    }
}

// ---------------- global mean pool over nodes ----------------
__global__ __launch_bounds__(128) void mean_pool(
    const float* __restrict__ H, float* __restrict__ HB)
{
    const int g = blockIdx.x, d = threadIdx.x;
    float s = 0.f;
    for (int n = 0; n < NNODE; ++n)
        s += H[((size_t)g * NNODE + n) * DMODEL + d];
    HB[g * DMODEL + d] = s * (1.f / NNODE);
}

// -------- linear time encoding: hin = hbar + dt*wt + bt (f32 + f16) --------
__global__ __launch_bounds__(128) void time_encode(
    const float* __restrict__ HB, const float* __restrict__ intervals,
    const float* __restrict__ wt, const float* __restrict__ bt,
    float* __restrict__ HIN, _Float16* __restrict__ HINH)
{
    const int i = blockIdx.x, d = threadIdx.x;
    const float v = HB[i * DMODEL + d] + intervals[i] * wt[d] + bt[d];
    HIN [i * DMODEL + d] = v;
    HINH[i * DMODEL + d] = (_Float16)v;
}

// -------- masked multi-head temporal attention (T=8, dh=32), f16 ctx --------
__global__ __launch_bounds__(512) void temporal_attn(
    const float* __restrict__ Q, const float* __restrict__ Km,
    const float* __restrict__ V, const int* __restrict__ seqlen,
    _Float16* __restrict__ CTX)
{
    const int tid = threadIdx.x;             // 512 = B * TH * T
    const int b = tid >> 5;
    const int h = (tid & 31) >> 3;
    const int q = tid & 7;
    int len = seqlen[b];
    if (len < 1) len = 1; if (len > TSEQ) len = TSEQ;
    const float scale = 0.17677669529663687f;   // 1/sqrt(32)

    float sc[TSEQ];
    float mx = -1e30f;
    for (int kt = 0; kt < TSEQ; ++kt) {
        float v = -1e9f;
        if (kt < len) {
            float s = 0.f;
            const float* qp = Q  + (size_t)(b * TSEQ + q)  * DMODEL + h * CHEAD;
            const float* kp = Km + (size_t)(b * TSEQ + kt) * DMODEL + h * CHEAD;
            for (int j = 0; j < CHEAD; ++j) s += qp[j] * kp[j];
            v = s * scale;
        }
        sc[kt] = v; mx = fmaxf(mx, v);
    }
    float dsum = 0.f;
    for (int kt = 0; kt < TSEQ; ++kt) { sc[kt] = __expf(sc[kt] - mx); dsum += sc[kt]; }
    const float inv = 1.f / dsum;
    for (int j = 0; j < CHEAD; ++j) {
        float c = 0.f;
        for (int kt = 0; kt < TSEQ; ++kt)
            c += sc[kt] * V[(size_t)(b * TSEQ + kt) * DMODEL + h * CHEAD + j];
        CTX[(size_t)(b * TSEQ + q) * DMODEL + h * CHEAD + j] = (_Float16)(c * inv);
    }
}

// ---------------- residual + masked mean + classifier ----------------
__global__ __launch_bounds__(128) void finalize(
    const float* __restrict__ HIN, const float* __restrict__ CTXO,
    const int* __restrict__ seqlen, const float* __restrict__ Wc,
    const float* __restrict__ bc, float* __restrict__ OUTP)
{
    __shared__ float summ[DMODEL];
    const int b = blockIdx.x, d = threadIdx.x;
    int len = seqlen[b];
    if (len < 1) len = 1; if (len > TSEQ) len = TSEQ;
    float s = 0.f;
    for (int t = 0; t < TSEQ; ++t)
        if (t < len)
            s += HIN [(size_t)(b * TSEQ + t) * DMODEL + d]
               + CTXO[(size_t)(b * TSEQ + t) * DMODEL + d];
    summ[d] = s / (float)len;
    __syncthreads();
    if (d < NCLS) {
        float o = bc[d];
        for (int j = 0; j < DMODEL; ++j) o += summ[j] * Wc[j * NCLS + d];
        OUTP[b * NCLS + d] = o;
    }
}

// ---------------------------------------------------------------------
extern "C" void kernel_launch(void* const* d_in, const int* in_sizes, int n_in,
                              void* d_out, int out_size, void* d_ws, size_t ws_size,
                              hipStream_t stream)
{
    const float* x_seq     = (const float*)d_in[0];
    const float* ea_seq    = (const float*)d_in[1];
    const float* intervals = (const float*)d_in[2];
    const int*   eidx      = (const int*)  d_in[3];
    const int*   seqlen    = (const int*)  d_in[4];
    const float* W0  = (const float*)d_in[5];
    const float* We0 = (const float*)d_in[6];
    const float* as0 = (const float*)d_in[7];
    const float* ad0 = (const float*)d_in[8];
    const float* ae0 = (const float*)d_in[9];
    const float* b0  = (const float*)d_in[10];
    const float* W1  = (const float*)d_in[11];
    const float* We1 = (const float*)d_in[12];
    const float* as1 = (const float*)d_in[13];
    const float* ad1 = (const float*)d_in[14];
    const float* ae1 = (const float*)d_in[15];
    const float* b1  = (const float*)d_in[16];
    const float* wt  = (const float*)d_in[17];
    const float* bt  = (const float*)d_in[18];
    const float* Wq  = (const float*)d_in[19];
    const float* Wk  = (const float*)d_in[20];
    const float* Wv  = (const float*)d_in[21];
    const float* Wo  = (const float*)d_in[22];
    const float* Wc  = (const float*)d_in[23];
    const float* bc  = (const float*)d_in[24];

    // ---- workspace carve-up (all regions 256B aligned) ----
    const size_t BIG  = (size_t)GCNT * NNODE * DMODEL;      // 8,388,608 elems
    const size_t XSZ  = (size_t)GCNT * NNODE * 64;          // x_seq elems
    char* base = (char*)d_ws;
    auto carve = [&](size_t bytes) {
        char* p = base;
        base += (bytes + 255) & ~(size_t)255;
        return p;
    };
    float*    HP   = (float*)   carve(BIG * 4);
    float*    H0   = (float*)   carve(BIG * 4);
    float*    H1   = (float*)   carve(BIG * 4);
    _Float16* XH   = (_Float16*)carve(XSZ * 2);
    _Float16* H0H  = (_Float16*)carve(BIG * 2);
    _Float16* WT0  = (_Float16*)carve((size_t)64  * DMODEL * 2);
    _Float16* WT1  = (_Float16*)carve((size_t)128 * DMODEL * 2);
    _Float16* WTq  = (_Float16*)carve((size_t)128 * DMODEL * 2);
    _Float16* WTk  = (_Float16*)carve((size_t)128 * DMODEL * 2);
    _Float16* WTv  = (_Float16*)carve((size_t)128 * DMODEL * 2);
    _Float16* WTo  = (_Float16*)carve((size_t)128 * DMODEL * 2);
    float*    HB   = (float*)   carve((size_t)GCNT * DMODEL * 4);
    float*    HIN  = (float*)   carve((size_t)GCNT * DMODEL * 4);
    _Float16* HINH = (_Float16*)carve((size_t)GCNT * DMODEL * 2);
    float*    Qb   = (float*)   carve((size_t)GCNT * DMODEL * 4);
    float*    Kb   = (float*)   carve((size_t)GCNT * DMODEL * 4);
    float*    Vb   = (float*)   carve((size_t)GCNT * DMODEL * 4);
    _Float16* CTH  = (_Float16*)carve((size_t)GCNT * DMODEL * 2);
    float*    CTO  = (float*)   carve((size_t)GCNT * DMODEL * 4);

    const dim3 blk128(128);

    // ---- precision prep: activations to f16, weights transposed to f16 ----
    f32_to_f16<<<dim3(2048), dim3(256), 0, stream>>>(x_seq, XH, (int)XSZ);
    wtrans<<<dim3(64),  blk128, 0, stream>>>(W0, WT0, 64);
    wtrans<<<dim3(128), blk128, 0, stream>>>(W1, WT1, 128);
    wtrans<<<dim3(128), blk128, 0, stream>>>(Wq, WTq, 128);
    wtrans<<<dim3(128), blk128, 0, stream>>>(Wk, WTk, 128);
    wtrans<<<dim3(128), blk128, 0, stream>>>(Wv, WTv, 128);
    wtrans<<<dim3(128), blk128, 0, stream>>>(Wo, WTo, 128);

    // ---- GAT layer 0 ----
    wmma_gemm128<<<dim3(GCNT, 8), blk128, 0, stream>>>(XH, WT0, HP, NNODE, 64);
    gat_attn<<<dim3(GCNT), dim3(512), 0, stream>>>(
        HP, ea_seq, eidx, We0, as0, ad0, ae0, b0, H0, H0H);

    // ---- GAT layer 1 ----
    wmma_gemm128<<<dim3(GCNT, 8), blk128, 0, stream>>>(H0H, WT1, HP, NNODE, 128);
    gat_attn<<<dim3(GCNT), dim3(512), 0, stream>>>(
        HP, ea_seq, eidx, We1, as1, ad1, ae1, b1, H1, (_Float16*)nullptr);

    // ---- temporal encoder ----
    mean_pool<<<dim3(GCNT), blk128, 0, stream>>>(H1, HB);
    time_encode<<<dim3(GCNT), blk128, 0, stream>>>(HB, intervals, wt, bt, HIN, HINH);

    wmma_gemm128<<<dim3(1, 2), blk128, 0, stream>>>(HINH, WTq, Qb, GCNT, DMODEL);
    wmma_gemm128<<<dim3(1, 2), blk128, 0, stream>>>(HINH, WTk, Kb, GCNT, DMODEL);
    wmma_gemm128<<<dim3(1, 2), blk128, 0, stream>>>(HINH, WTv, Vb, GCNT, DMODEL);

    temporal_attn<<<dim3(1), dim3(512), 0, stream>>>(Qb, Kb, Vb, seqlen, CTH);

    wmma_gemm128<<<dim3(1, 2), blk128, 0, stream>>>(CTH, WTo, CTO, GCNT, DMODEL);
    finalize<<<dim3(BATCH), blk128, 0, stream>>>(HIN, CTO, seqlen, Wc, bc,
                                                 (float*)d_out);
}
